// PatchSelfAttentionReg_76029511074040
// MI455X (gfx1250) — compile-verified, matrix operand order checked by reading, not run on hardware
//
#include <hip/hip_runtime.h>

typedef _Float16 half_t;
typedef __attribute__((ext_vector_type(16))) _Float16 v16h;
typedef __attribute__((ext_vector_type(8)))  _Float16 v8h;
typedef __attribute__((ext_vector_type(8)))  float    v8f;
typedef __attribute__((ext_vector_type(4)))  float    v4f;

#define DIMC     256
#define HEADS    8
#define HEAD_DIM 32
#define T_TOK    1600
#define N_QK     512                 // q rows [0,256) + k rows [256,512) of qkv_w
#define ATT_SCALE 0.17677669529663687f   // 32^-0.5
#define LN_EPS   1e-6f

// ---------------------------------------------------------------------------
// Kernel 0: convert first 512 rows of qkv_w (f32) -> f16  (512*256 elements)
// ---------------------------------------------------------------------------
__global__ __launch_bounds__(256) void wconv_kernel(const float* __restrict__ w,
                                                    half_t* __restrict__ wh) {
    int i = blockIdx.x * 256 + threadIdx.x;     // grid = 512 blocks
    wh[i] = (half_t)w[i];
}

// ---------------------------------------------------------------------------
// Kernel 1: LayerNorm + QK projection for a 16-token tile of one batch.
//   grid = (T/16, B), block = 256 (8 waves)
//   A (tokens, f16 after LN) lives in LDS; B = f16 weights from L2.
//   GEMM 16 x 512, K=256 via v_wmma_f32_16x16x32_f16.
// ---------------------------------------------------------------------------
#define XS_P 272   // f32 pitch (272 % 64 == 16 -> conflict-free half-wave rows)
#define LN_P 264   // f16 pitch (row start stays 16B aligned: 264*2 = 33*16)

__global__ __launch_bounds__(256) void ln_qk_kernel(
    const float* __restrict__ feat, const float* __restrict__ lnw,
    const float* __restrict__ lnb,  const half_t* __restrict__ wqk,
    const float* __restrict__ qkvb, half_t* __restrict__ qh,
    half_t* __restrict__ kh) {

    const int b    = blockIdx.y;
    const int t0   = blockIdx.x * 16;
    const int tid  = threadIdx.x;
    const int lane = tid & 31;
    const int wave = tid >> 5;
    const int hi16 = lane >> 4;      // which half of the wave
    const int m    = lane & 15;      // matrix row/col owned by this lane

    __shared__ float  xs[16][XS_P];      // token-major f32 tile (transposed feat)
    __shared__ half_t lnA[16][LN_P];     // LayerNormed tokens, f16

    // ---- load feat: thread tid owns channel c = tid, 16 contiguous tokens ----
    {
        const float* src = feat + ((size_t)b * DIMC + tid) * T_TOK + t0;
        v4f v0 = ((const v4f*)src)[0];
        v4f v1 = ((const v4f*)src)[1];
        v4f v2 = ((const v4f*)src)[2];
        v4f v3 = ((const v4f*)src)[3];
        xs[ 0][tid] = v0[0]; xs[ 1][tid] = v0[1]; xs[ 2][tid] = v0[2]; xs[ 3][tid] = v0[3];
        xs[ 4][tid] = v1[0]; xs[ 5][tid] = v1[1]; xs[ 6][tid] = v1[2]; xs[ 7][tid] = v1[3];
        xs[ 8][tid] = v2[0]; xs[ 9][tid] = v2[1]; xs[10][tid] = v2[2]; xs[11][tid] = v2[3];
        xs[12][tid] = v3[0]; xs[13][tid] = v3[1]; xs[14][tid] = v3[2]; xs[15][tid] = v3[3];
    }
    __syncthreads();

    // ---- LayerNorm: 16 lanes per token, shuffle-reduce within 16-lane group ----
    {
        const int row = tid >> 4;        // token 0..15
        const int j   = tid & 15;
        float s = 0.f, s2 = 0.f;
        #pragma unroll
        for (int it = 0; it < 16; ++it) {
            float x = xs[row][j + 16 * it];
            s += x; s2 += x * x;
        }
        #pragma unroll
        for (int msk = 8; msk; msk >>= 1) {
            s  += __shfl_xor(s,  msk, 32);
            s2 += __shfl_xor(s2, msk, 32);
        }
        float mu   = s * (1.f / 256.f);
        float var  = s2 * (1.f / 256.f) - mu * mu;
        float rstd = rsqrtf(var + LN_EPS);
        #pragma unroll
        for (int it = 0; it < 16; ++it) {
            int c = j + 16 * it;
            float y = (xs[row][c] - mu) * rstd * lnw[c] + lnb[c];
            lnA[row][c] = (half_t)y;
        }
    }
    __syncthreads();

    // ---- GEMM: 32 N-tiles of 16 outputs; each wave owns 4 tiles ----
    for (int nt = wave * 4; nt < wave * 4 + 4; ++nt) {
        const int n = nt * 16 + m;                      // output feature of this lane
        const half_t* brow = wqk + (size_t)n * DIMC;    // weight row n (f16)
        v8f acc = {};
        #pragma unroll
        for (int kk = 0; kk < 8; ++kk) {
            const int k0 = kk * 32 + hi16 * 8;
            v8h alo = *(const v8h*)&lnA[m][k0];
            v8h ahi = *(const v8h*)&lnA[m][k0 + 16];
            v16h a = __builtin_shufflevector(alo, ahi,
                     0,1,2,3,4,5,6,7,8,9,10,11,12,13,14,15);
            v8h blo = *(const v8h*)(brow + k0);
            v8h bhi = *(const v8h*)(brow + k0 + 16);
            v16h bmat = __builtin_shufflevector(blo, bhi,
                     0,1,2,3,4,5,6,7,8,9,10,11,12,13,14,15);
            acc = __builtin_amdgcn_wmma_f32_16x16x32_f16(
                      false, a, false, bmat, (short)0, acc, false, false);
        }
        const float bias = qkvb[n];
        half_t* base = (n < 256) ? qh : kh;
        const int nn = n & 255;
        const int h  = nn >> 5;
        const int d  = nn & 31;
        const size_t rowbase =
            ((size_t)(b * HEADS + h) * T_TOK + t0) * HEAD_DIM + d;
        #pragma unroll
        for (int r = 0; r < 8; ++r) {
            int mm = r + 8 * hi16;                       // token within tile
            base[rowbase + (size_t)mm * HEAD_DIM] = (half_t)(acc[r] + bias);
        }
    }
}

// ---------------------------------------------------------------------------
// Kernel 2: fused scores + softmax for one (b, h, 16-row stripe).
//   grid = (100, 8, 8), block = 256 (8 waves).
//   One v_wmma_f32_16x16x32_f16 per 16x16 score tile (d=32 = full K).
//   Stripe of 16 x 1600 f32 scores staged in LDS (pitch 1604), softmax with
//   b128 LDS traffic, then a single coalesced NON-TEMPORAL write of attn
//   (655 MB stream must not thrash the L2 that keeps K resident).
// ---------------------------------------------------------------------------
#define SC_P 1604                                   // 1604 % 64 == 4; 16B-aligned rows
#define K2_SMEM (16 * SC_P * 4 + 16 * 4)

__global__ __launch_bounds__(256) void attn_kernel(
    const half_t* __restrict__ qh, const half_t* __restrict__ kh,
    float* __restrict__ out) {

    const int mt   = blockIdx.x;      // row-tile 0..99
    const int h    = blockIdx.y;
    const int b    = blockIdx.z;
    const int tid  = threadIdx.x;
    const int lane = tid & 31;
    const int wave = tid >> 5;
    const int hi16 = lane >> 4;
    const int m    = lane & 15;

    extern __shared__ char smem[];
    float* sc   = (float*)smem;                     // [16][SC_P]
    float* rinv = (float*)(smem + 16 * SC_P * 4);   // [16]

    const size_t bh    = (size_t)(b * HEADS + h);
    const half_t* qbase = qh + (bh * T_TOK + (size_t)mt * 16) * HEAD_DIM;
    const half_t* kbase = kh +  bh * T_TOK * HEAD_DIM;

    // A fragment: q rows, loaded once per wave (reused for all column tiles)
    v16h afrag;
    {
        const half_t* qrow = qbase + (size_t)m * HEAD_DIM + hi16 * 8;
        v8h lo = *(const v8h*)(qrow);
        v8h hi = *(const v8h*)(qrow + 16);
        afrag = __builtin_shufflevector(lo, hi,
                 0,1,2,3,4,5,6,7,8,9,10,11,12,13,14,15);
    }

    // Column-tile sweep: 100 tiles round-robin over 8 waves
    for (int nt = wave; nt < 100; nt += 8) {
        const half_t* krow = kbase + (size_t)(nt * 16 + m) * HEAD_DIM + hi16 * 8;
        // cover the next tile's B-fragment latency (global_prefetch_b8)
        if (nt + 8 < 100)
            __builtin_prefetch(krow + 8 * 16 * HEAD_DIM, 0, 3);
        v8h lo = *(const v8h*)(krow);
        v8h hi = *(const v8h*)(krow + 16);
        v16h bfrag = __builtin_shufflevector(lo, hi,
                 0,1,2,3,4,5,6,7,8,9,10,11,12,13,14,15);
        v8f acc = {};
        acc = __builtin_amdgcn_wmma_f32_16x16x32_f16(
                  false, afrag, false, bfrag, (short)0, acc, false, false);
        const int col = nt * 16 + m;
        #pragma unroll
        for (int r = 0; r < 8; ++r)
            sc[(r + 8 * hi16) * SC_P + col] = acc[r] * ATT_SCALE;
    }
    __syncthreads();

    // Row softmax: 16 lanes per row, b128 LDS traffic (25 float4 per lane)
    {
        const int row = tid >> 4;
        const int j   = tid & 15;
        v4f* srow4 = (v4f*)(sc + row * SC_P);
        float mx = -3.4e38f;
        #pragma unroll 5
        for (int it = 0; it < 25; ++it) {
            v4f v = srow4[j + 16 * it];
            mx = fmaxf(mx, fmaxf(fmaxf(v[0], v[1]), fmaxf(v[2], v[3])));
        }
        #pragma unroll
        for (int msk = 8; msk; msk >>= 1)
            mx = fmaxf(mx, __shfl_xor(mx, msk, 32));
        float sum = 0.f;
        #pragma unroll 5
        for (int it = 0; it < 25; ++it) {
            v4f v = srow4[j + 16 * it];
            v[0] = __expf(v[0] - mx);
            v[1] = __expf(v[1] - mx);
            v[2] = __expf(v[2] - mx);
            v[3] = __expf(v[3] - mx);
            srow4[j + 16 * it] = v;
            sum += (v[0] + v[1]) + (v[2] + v[3]);
        }
        #pragma unroll
        for (int msk = 8; msk; msk >>= 1)
            sum += __shfl_xor(sum, msk, 32);
        if (j == 0) rinv[row] = 1.f / sum;
    }
    __syncthreads();

    // Coalesced non-temporal streaming write of the 16 x 1600 stripe
    const size_t obase = (bh * T_TOK + (size_t)mt * 16) * T_TOK;
    for (int rr = 0; rr < 16; ++rr) {
        const float scale = rinv[rr];
        const v4f* s4 = (const v4f*)(sc + rr * SC_P);
        v4f* o4 = (v4f*)(out + obase + (size_t)rr * T_TOK);
        for (int i = tid; i < T_TOK / 4; i += 256) {     // 400 float4 per row
            v4f v = s4[i];
            v[0] *= scale; v[1] *= scale; v[2] *= scale; v[3] *= scale;
            __builtin_nontemporal_store(v, o4 + i);
        }
    }
}

// ---------------------------------------------------------------------------
extern "C" void kernel_launch(void* const* d_in, const int* in_sizes, int n_in,
                              void* d_out, int out_size, void* d_ws, size_t ws_size,
                              hipStream_t stream) {
    const float* feat = (const float*)d_in[0];
    const float* lnw  = (const float*)d_in[1];
    const float* lnb  = (const float*)d_in[2];
    const float* qkvw = (const float*)d_in[3];
    const float* qkvb = (const float*)d_in[4];
    float* out = (float*)d_out;

    char* ws = (char*)d_ws;
    half_t* wqk = (half_t*)ws;                                   // 512*256*2 B
    const size_t wqk_bytes = (size_t)N_QK * DIMC * sizeof(half_t);
    const size_t qk_elems  = (size_t)8 * HEADS * T_TOK * HEAD_DIM;
    half_t* qh = (half_t*)(ws + wqk_bytes);                      // 6.55 MB
    half_t* kh = (half_t*)(ws + wqk_bytes + qk_elems * sizeof(half_t));

    // allow >64KB dynamic LDS for the fused attention kernel
    hipFuncSetAttribute((const void*)attn_kernel,
                        hipFuncAttributeMaxDynamicSharedMemorySize, K2_SMEM);

    wconv_kernel<<<N_QK, 256, 0, stream>>>(qkvw, wqk);
    ln_qk_kernel<<<dim3(T_TOK / 16, 8), 256, 0, stream>>>(
        feat, lnw, lnb, wqk, qkvb, qh, kh);
    attn_kernel<<<dim3(T_TOK / 16, HEADS, 8), 256, K2_SMEM, stream>>>(qh, kh, out);
}
